// TimeLSTM_16346645528871
// MI455X (gfx1250) — matrix operand order repaired and checked by reading, not compiled, not run on hardware
//
#include <hip/hip_runtime.h>
#include <stdint.h>

// ---------------------------------------------------------------------------
// Time-LSTM for MI455X (gfx1250): persistent wave32 WMMA-bf16 kernel.
// gates = [h | x_s] @ [W_all ; U_all]  (K = 768), c_s1 = tanh(c @ W_d).
// ---------------------------------------------------------------------------

#define BB 64
#define SS 512
#define DD 256
#define HH 512
#define NBLK 16   // persistent blocks (must all be co-resident)

typedef __attribute__((ext_vector_type(16))) __bf16 v16bf;
typedef __attribute__((ext_vector_type(8)))  float  v8f;

union Frag16 {
  v16bf    v;
  uint32_t u32[8];
  uint4    q[2];
};

__device__ __forceinline__ uint16_t f2bf(float x) {
  uint32_t u = __float_as_uint(x);
  uint32_t r = u + 0x7FFFu + ((u >> 16) & 1u);   // round-to-nearest-even
  return (uint16_t)(r >> 16);
}

__device__ __forceinline__ float sigmoidf_fast(float x) {
  return 1.0f / (1.0f + __expf(-x));
}

// A-fragment (16x32 bf16, MxK) from an LDS tile of shape [16, stride].
// ISA layout: lanes 0-15 -> M=L, K in {0..7,16..23}; lanes 16-31 -> K+8.
// VGPR j2 holds the contiguous K-pair starting at (j2<4 ? 2*j2 : 8+2*j2).
__device__ __forceinline__ Frag16 load_a_lds(const uint16_t* tile, int stride,
                                             int kbase, int lane) {
  Frag16 a;
  const uint16_t* row = tile + (lane & 15) * stride + kbase + ((lane & 16) ? 8 : 0);
#pragma unroll
  for (int j2 = 0; j2 < 8; ++j2) {
    int kk = (j2 < 4) ? (2 * j2) : (8 + 2 * j2);
    a.u32[j2] = *(const uint32_t*)(row + kk);
  }
  return a;
}

// B-fragment (32x16 bf16, KxN) from pre-swizzled global weights:
// fragment is 32 lanes x 16 bf16 contiguous -> 2x B128 per lane.
__device__ __forceinline__ Frag16 load_b_glb(const uint16_t* base, int frag, int lane) {
  Frag16 b;
  const uint4* p = (const uint4*)(base + ((size_t)frag << 9) + (lane << 4));
  b.q[0] = p[0];
  b.q[1] = p[1];
  return b;
}

__device__ __forceinline__ v8f wmma_bf16(const Frag16& a, const Frag16& b, v8f c) {
  return __builtin_amdgcn_wmma_f32_16x16x32_bf16(false, a.v, false, b.v,
                                                 (short)0, c, false, false);
}

// ---------------------------------------------------------------------------
// Prep kernels
// ---------------------------------------------------------------------------

__global__ void convert_inputs(const float* __restrict__ x,
                               uint16_t* __restrict__ xb, int n) {
  int i = (blockIdx.x * 256 + threadIdx.x) * 4;
  if (i < n) {
    float4 v = *(const float4*)(x + i);
    xb[i + 0] = f2bf(v.x);
    xb[i + 1] = f2bf(v.y);
    xb[i + 2] = f2bf(v.z);
    xb[i + 3] = f2bf(v.w);
  }
}

__global__ void init_state(uint16_t* __restrict__ hbf,
                           uint16_t* __restrict__ cbf,
                           float* __restrict__ cf) {
  int i = blockIdx.x * 256 + threadIdx.x;   // 32768 threads
  hbf[i] = 0;
  cbf[i] = 0;
  cf[i]  = 0.0f;
}

// Pre-swizzle weights into WMMA-B fragment-contiguous bf16 layout.
// Wg: combined [W_all(512 rows); U_all(256 rows)] x 2048 cols ->
//     24 kTiles x 128 nTiles fragments. Wd: 16 x 32 fragments.
// Fragment element j of lane L = W[k = kt*32 + (L>=16?8:0) + j + (j>=8?8:0)][ntile*16 + L%16].
__global__ void swizzle_weights(const float* __restrict__ W_all,
                                const float* __restrict__ U_all,
                                const float* __restrict__ W_d,
                                uint16_t* __restrict__ Wg,
                                uint16_t* __restrict__ Wdw) {
  int t = blockIdx.x * 256 + threadIdx.x;
  if (t < 24 * 128 * 32) {
    int frag = t >> 5, lane = t & 31;
    int kt = frag >> 7, ntile = frag & 127;
    int n = ntile * 16 + (lane & 15);
    int kb = kt * 32 + ((lane & 16) ? 8 : 0);
    uint16_t* dst = Wg + ((size_t)frag << 9) + (lane << 4);
#pragma unroll
    for (int j = 0; j < 16; ++j) {
      int k = kb + j + ((j >= 8) ? 8 : 0);
      float v = (k < HH) ? W_all[k * (4 * HH) + n] : U_all[(k - HH) * (4 * HH) + n];
      dst[j] = f2bf(v);
    }
  } else {
    t -= 24 * 128 * 32;
    int frag = t >> 5, lane = t & 31;
    int kt = frag >> 5, ntile = frag & 31;
    int n = ntile * 16 + (lane & 15);
    int kb = kt * 32 + ((lane & 16) ? 8 : 0);
    uint16_t* dst = Wdw + ((size_t)frag << 9) + (lane << 4);
#pragma unroll
    for (int j = 0; j < 16; ++j) dst[j] = f2bf(W_d[(kb + j + ((j >= 8) ? 8 : 0)) * HH + n]);
  }
}

// ---------------------------------------------------------------------------
// Persistent recurrent kernel: 16 blocks x 256 threads (8 wave32 / block).
// unit = block*8 + wave in [0,128): mt = unit/32 (batch tile), nt = unit%32.
// All waves in a block share mt -> h/c/x row tiles staged once in LDS.
// One atomic global barrier per time step (monotonic counter, double-buffered
// state -> single barrier is race-free).
// ---------------------------------------------------------------------------

__global__ __launch_bounds__(256, 1)
void time_lstm_persistent(const float* __restrict__ ts,
                          const uint16_t* __restrict__ Wg,
                          const uint16_t* __restrict__ Wd,
                          const uint16_t* __restrict__ Xbf,
                          uint16_t* __restrict__ hbf,   // 2 x [64,512] bf16
                          uint16_t* __restrict__ cbf,   // 2 x [64,512] bf16
                          float* __restrict__ cf,       // 2 x [64,512] f32
                          float* __restrict__ out,      // [B,S,H] ++ h ++ c
                          uint32_t* __restrict__ bar) {
  __shared__ __align__(16) uint16_t hT[16 * HH];   // 16 KB
  __shared__ __align__(16) uint16_t cT[16 * HH];   // 16 KB
  __shared__ __align__(16) uint16_t xT[16 * DD];   //  8 KB

  const int tid  = threadIdx.x;
  const int lane = tid & 31;
  const int wave = tid >> 5;
  const int unit = blockIdx.x * 8 + wave;  // 0..127
  const int mt   = unit >> 5;              // 0..3
  const int nt   = unit & 31;              // 0..31
  const int ncol = lane & 15;
  const int mrow = (lane & 16) ? 8 : 0;
  const int colH = nt * 16 + ncol;

  for (int s = 0; s < SS; ++s) {
    const int rd = s & 1, wr = rd ^ 1;
    const uint16_t* hIn  = hbf + rd * (BB * HH);
    const uint16_t* cIn  = cbf + rd * (BB * HH);
    const float*    cfIn = cf  + rd * (BB * HH);
    uint16_t* hOut  = hbf + wr * (BB * HH);
    uint16_t* cOut  = cbf + wr * (BB * HH);
    float*    cfOut = cf  + wr * (BB * HH);

    // ---- stage this block's 16 batch rows of h, c (bf16) and x_s into LDS
    {
      const uint4* srcH = (const uint4*)(hIn + mt * 16 * HH);
      const uint4* srcC = (const uint4*)(cIn + mt * 16 * HH);
      uint4* dH = (uint4*)hT;
      uint4* dC = (uint4*)cT;
#pragma unroll
      for (int i = tid; i < 1024; i += 256) { dH[i] = srcH[i]; dC[i] = srcC[i]; }
      uint4* dX = (uint4*)xT;
      for (int i = tid; i < 512; i += 256) {
        int r = i >> 5, part = i & 31;
        const uint4* srcX =
            (const uint4*)(Xbf + ((size_t)(mt * 16 + r) * SS + s) * DD);
        dX[i] = srcX[part];
      }
    }
    __syncthreads();

    // ---- GEMMs: gates(f,i,o,ctmp) over K=768 ([h|x]), c@W_d over K=512
    v8f accF = {}, accI = {}, accO = {}, accC = {}, accD = {};

#pragma unroll 4
    for (int kt = 0; kt < 16; ++kt) {
      Frag16 aH = load_a_lds(hT, HH, kt * 32, lane);
      Frag16 aC = load_a_lds(cT, HH, kt * 32, lane);
      if (kt < 15)
        __builtin_prefetch(Wg + ((size_t)((kt + 1) * 128 + nt) << 9) + (lane << 4), 0, 3);
      Frag16 b0 = load_b_glb(Wg, kt * 128 + nt,      lane);
      Frag16 b1 = load_b_glb(Wg, kt * 128 + 32 + nt, lane);
      Frag16 b2 = load_b_glb(Wg, kt * 128 + 64 + nt, lane);
      Frag16 b3 = load_b_glb(Wg, kt * 128 + 96 + nt, lane);
      Frag16 bd = load_b_glb(Wd, kt * 32 + nt, lane);
      accF = wmma_bf16(aH, b0, accF);
      accI = wmma_bf16(aH, b1, accI);
      accO = wmma_bf16(aH, b2, accO);
      accC = wmma_bf16(aH, b3, accC);
      accD = wmma_bf16(aC, bd, accD);
    }
#pragma unroll 4
    for (int kt = 0; kt < 8; ++kt) {
      Frag16 aX = load_a_lds(xT, DD, kt * 32, lane);
      int fb = (16 + kt) * 128 + nt;
      Frag16 b0 = load_b_glb(Wg, fb,      lane);
      Frag16 b1 = load_b_glb(Wg, fb + 32, lane);
      Frag16 b2 = load_b_glb(Wg, fb + 64, lane);
      Frag16 b3 = load_b_glb(Wg, fb + 96, lane);
      accF = wmma_bf16(aX, b0, accF);
      accI = wmma_bf16(aX, b1, accI);
      accO = wmma_bf16(aX, b2, accO);
      accC = wmma_bf16(aX, b3, accC);
    }

    // ---- elementwise epilogue (wave-local; C layout: VGPR r -> M=r+mrow)
#pragma unroll
    for (int r = 0; r < 8; ++r) {
      int b = mt * 16 + mrow + r;
      float t    = ts[b * SS + s];
      float cs1  = tanhf(accD[r]);
      float cold = cfIn[b * HH + colH];
      float cadj = cold - cs1 + cs1 * t;
      float fg = sigmoidf_fast(accF[r]);
      float ig = sigmoidf_fast(accI[r]);
      float og = sigmoidf_fast(accO[r]);
      float cg = sigmoidf_fast(accC[r]);
      float cnew = fg * cadj + ig * cg;
      float hnew = og * tanhf(cnew);
      out[((size_t)b * SS + s) * HH + colH] = og;   // reference emits gate o
      cfOut[b * HH + colH] = cnew;
      cOut[b * HH + colH]  = f2bf(cnew);
      hOut[b * HH + colH]  = f2bf(hnew);
      if (s == SS - 1) {
        out[(size_t)BB * SS * HH + b * HH + colH]            = hnew;
        out[(size_t)BB * SS * HH + BB * HH + b * HH + colH]  = cnew;
      }
    }

    // ---- device-wide barrier (release -> arrive -> spin -> acquire)
    __threadfence();
    __syncthreads();
    if (tid == 0) {
      atomicAdd(bar, 1u);
      uint32_t target = (uint32_t)NBLK * (uint32_t)(s + 1);
      volatile uint32_t* vb = (volatile uint32_t*)bar;
      while (*vb < target) __builtin_amdgcn_s_sleep(1);
    }
    __syncthreads();
    __threadfence();
  }
}

// ---------------------------------------------------------------------------
// Launch
// ---------------------------------------------------------------------------

extern "C" void kernel_launch(void* const* d_in, const int* in_sizes, int n_in,
                              void* d_out, int out_size, void* d_ws, size_t ws_size,
                              hipStream_t stream) {
  const float* inputs = (const float*)d_in[0];  // [B,S,D]
  const float* tstamp = (const float*)d_in[1];  // [B,S]
  const float* W_all  = (const float*)d_in[2];  // [H,4H]
  const float* U_all  = (const float*)d_in[3];  // [D,4H]
  const float* W_d    = (const float*)d_in[4];  // [H,H]
  float* out = (float*)d_out;

  char* ws = (char*)d_ws;
  size_t off = 0;
  uint32_t* bar = (uint32_t*)(ws + off); off += 256;
  uint16_t* Wg  = (uint16_t*)(ws + off); off += (size_t)24 * 128 * 512 * 2;  // 3 MB
  uint16_t* Wdw = (uint16_t*)(ws + off); off += (size_t)16 * 32 * 512 * 2;   // 0.5 MB
  uint16_t* Xbf = (uint16_t*)(ws + off); off += (size_t)BB * SS * DD * 2;    // 16 MB
  uint16_t* hbf = (uint16_t*)(ws + off); off += (size_t)2 * BB * HH * 2;
  uint16_t* cbf = (uint16_t*)(ws + off); off += (size_t)2 * BB * HH * 2;
  float*    cf  = (float*)(ws + off);    off += (size_t)2 * BB * HH * 4;
  (void)off; (void)ws_size; (void)in_sizes; (void)n_in; (void)out_size;

  (void)hipMemsetAsync(bar, 0, 256, stream);
  convert_inputs<<<(BB * SS * DD / 4 + 255) / 256, 256, 0, stream>>>(
      inputs, Xbf, BB * SS * DD);
  init_state<<<(BB * HH) / 256, 256, 0, stream>>>(hbf, cbf, cf);
  swizzle_weights<<<(24 * 128 * 32 + 16 * 32 * 32) / 256, 256, 0, stream>>>(
      W_all, U_all, W_d, Wg, Wdw);
  time_lstm_persistent<<<NBLK, 256, 0, stream>>>(
      tstamp, Wg, Wdw, Xbf, hbf, cbf, cf, out, bar);
}
